// MultiHeadAttention_26757646254489
// MI455X (gfx1250) — compile-verified, matrix operand order checked by reading, not run on hardware
//
#include <hip/hip_runtime.h>
#include <hip/hip_bf16.h>

typedef __attribute__((ext_vector_type(16))) _Float16 v16h;
typedef __attribute__((ext_vector_type(8)))  _Float16 v8h;
typedef __attribute__((ext_vector_type(8)))  float    v8f;

constexpr int Bsz  = 4;
constexpr int Seq  = 2048;
constexpr int Emb  = 1024;
constexpr int Hn   = 16;
constexpr int Dh   = 64;
constexpr int Rows = Bsz * Seq;   // 8192

// ---------------------------------------------------------------------------
// Fragment helpers (CDNA5 wave32 WMMA, 16x16x32 f16 -> f32)
// A 16x32 f16 layout (ISA 7.12.2): lane L(0..15) row M=L holds K 0..7,16..23;
// lane L+16 holds K 8..15,24..31. B fragments use the same pattern on rows of
// B^T, so one K-contiguous loader (2 x 128-bit) serves both operands.
// ---------------------------------------------------------------------------
__device__ __forceinline__ v16h load_frag(const _Float16* p0, int ld, int lane) {
  const int lo = lane & 15, hi = lane >> 4;
  const _Float16* p = p0 + (size_t)lo * ld + hi * 8;
  v8h x0 = *(const v8h*)(p);
  v8h x1 = *(const v8h*)(p + 16);
  return __builtin_shufflevector(x0, x1, 0,1,2,3,4,5,6,7,8,9,10,11,12,13,14,15);
}

__device__ __forceinline__ v8f wmma16(v16h a, v16h b, v8f c) {
  return __builtin_amdgcn_wmma_f32_16x16x32_f16(false, a, false, b, (short)0, c,
                                                false, false);
}

// ---------------------------------------------------------------------------
// Conversions / repacks (bandwidth trivial, run once per launch)
// ---------------------------------------------------------------------------
__global__ void cvt_x_kernel(const float* __restrict__ x, _Float16* __restrict__ xh) {
  size_t i = (size_t)blockIdx.x * 256 + threadIdx.x;
  xh[i] = (_Float16)x[i];
}

// Wt[p][h][d][e] = W_p[h][e][d]
__global__ void cvt_wqkv_kernel(const float* __restrict__ Wq, const float* __restrict__ Wk,
                                const float* __restrict__ Wv, _Float16* __restrict__ Wt) {
  size_t i = (size_t)blockIdx.x * 256 + threadIdx.x;   // over 3*Hn*Dh*Emb
  int e = (int)(i % Emb);
  int d = (int)((i / Emb) % Dh);
  int h = (int)((i / ((size_t)Emb * Dh)) % Hn);
  int p = (int)( i / ((size_t)Emb * Dh * Hn));
  const float* W = (p == 0) ? Wq : (p == 1) ? Wk : Wv;
  Wt[i] = (_Float16)W[((size_t)h * Emb + e) * Dh + d];
}

// Wot[n][k] = Wo[k][n]
__global__ void cvt_wo_kernel(const float* __restrict__ Wo, _Float16* __restrict__ Wot) {
  size_t i = (size_t)blockIdx.x * 256 + threadIdx.x;   // over Emb*Emb
  int k = (int)(i % Emb);
  int n = (int)(i / Emb);
  Wot[i] = (_Float16)Wo[(size_t)k * Emb + n];
}

// ---------------------------------------------------------------------------
// Fused QKV projection.  grid (Rows/128, Hn, 3), block 128.
// Each wave computes a 32x64 tile (2 A-frags x 4 B-frags -> 8 WMMA / chunk),
// with register double-buffering of the next K-chunk so global loads overlap
// the WMMA chain instead of serializing on s_wait_loadcnt 0.
// ---------------------------------------------------------------------------
__global__ void gemm_qkv_kernel(const _Float16* __restrict__ xh,
                                const _Float16* __restrict__ Wt,
                                const float* __restrict__ bq, const float* __restrict__ bk,
                                const float* __restrict__ bv,
                                _Float16* __restrict__ Qf, _Float16* __restrict__ Kf,
                                _Float16* __restrict__ Vt) {
  const int mt = blockIdx.x, h = blockIdx.y, p = blockIdx.z;
  const int wave = threadIdx.x >> 5, lane = threadIdx.x & 31;
  const int lo = lane & 15, hi = lane >> 4;
  const int m0 = mt * 128 + wave * 32;

  const _Float16* A0 = xh + (size_t)m0 * Emb;
  const _Float16* A1 = A0 + (size_t)16 * Emb;
  const _Float16* Bm = Wt + ((size_t)(p * Hn + h) * Dh) * Emb;   // [d][e], ld=Emb

  v16h a0 = load_frag(A0, Emb, lane);
  v16h a1 = load_frag(A1, Emb, lane);
  v16h b0 = load_frag(Bm + (size_t) 0 * Emb, Emb, lane);
  v16h b1 = load_frag(Bm + (size_t)16 * Emb, Emb, lane);
  v16h b2 = load_frag(Bm + (size_t)32 * Emb, Emb, lane);
  v16h b3 = load_frag(Bm + (size_t)48 * Emb, Emb, lane);

  v8f acc[2][4] = {};
  for (int k0 = 0; k0 < Emb; k0 += 32) {
    const int kn = (k0 + 32) & (Emb - 1);       // branchless wrap (last iter redundant)
    v16h a0n = load_frag(A0 + kn, Emb, lane);
    v16h a1n = load_frag(A1 + kn, Emb, lane);
    v16h bn0 = load_frag(Bm + (size_t) 0 * Emb + kn, Emb, lane);
    v16h bn1 = load_frag(Bm + (size_t)16 * Emb + kn, Emb, lane);
    v16h bn2 = load_frag(Bm + (size_t)32 * Emb + kn, Emb, lane);
    v16h bn3 = load_frag(Bm + (size_t)48 * Emb + kn, Emb, lane);

    acc[0][0] = wmma16(a0, b0, acc[0][0]);
    acc[1][0] = wmma16(a1, b0, acc[1][0]);
    acc[0][1] = wmma16(a0, b1, acc[0][1]);
    acc[1][1] = wmma16(a1, b1, acc[1][1]);
    acc[0][2] = wmma16(a0, b2, acc[0][2]);
    acc[1][2] = wmma16(a1, b2, acc[1][2]);
    acc[0][3] = wmma16(a0, b3, acc[0][3]);
    acc[1][3] = wmma16(a1, b3, acc[1][3]);

    a0 = a0n; a1 = a1n; b0 = bn0; b1 = bn1; b2 = bn2; b3 = bn3;
  }

  const float* bias = (p == 0) ? bq : (p == 1) ? bk : bv;
  const float  csc  = (p == 0) ? (0.125f * 1.44269504f) : 1.0f;

  #pragma unroll
  for (int mi = 0; mi < 2; ++mi) {
    #pragma unroll
    for (int i = 0; i < 8; ++i) {
      const int row = m0 + mi * 16 + hi * 8 + i;
      const int bb  = row / Seq, ss = row % Seq;
      #pragma unroll
      for (int n = 0; n < 4; ++n) {
        const int d = n * 16 + lo;
        const float val = (acc[mi][n][i] + bias[h * Dh + d]) * csc;
        if (p == 2) {
          Vt[(((size_t)(h * Bsz + bb) * Dh) + d) * Seq + ss] = (_Float16)val;
        } else {
          _Float16* dst = (p == 0) ? Qf : Kf;
          dst[(((size_t)(h * Bsz + bb) * Seq) + ss) * Dh + d] = (_Float16)val;
        }
      }
    }
  }
}

// ---------------------------------------------------------------------------
// Flash attention: grid (Seq/64, Hn, Bsz), block 128 (4 waves x 16 q-rows).
// K/V tiles (shared by all 4 waves) are staged in LDS once per block;
// next tile is prefetched (global_prefetch_b8) during compute.
// ---------------------------------------------------------------------------
__global__ void attn_kernel(const _Float16* __restrict__ Qf, const _Float16* __restrict__ Kf,
                            const _Float16* __restrict__ Vt, _Float16* __restrict__ ctxh) {
  const int qb = blockIdx.x, h = blockIdx.y, b = blockIdx.z;
  const int tid  = threadIdx.x;
  const int wave = tid >> 5, lane = tid & 31;
  const int lo = lane & 15, hi = lane >> 4;
  const int q0 = qb * 64 + wave * 16;                      // sequence-local

  const _Float16* Qb = Qf + ((size_t)(h * Bsz + b) * Seq) * Dh;
  const _Float16* Kb = Kf + ((size_t)(h * Bsz + b) * Seq) * Dh;
  const _Float16* Vb = Vt + ((size_t)(h * Bsz + b) * Dh) * Seq;

  const v16h aq0 = load_frag(Qb + (size_t)q0 * Dh +  0, Dh, lane);
  const v16h aq1 = load_frag(Qb + (size_t)q0 * Dh + 32, Dh, lane);

  v8f acc[4] = {v8f{}, v8f{}, v8f{}, v8f{}};
  float mrow[8], lrow[8];
  #pragma unroll
  for (int i = 0; i < 8; ++i) { mrow[i] = -1e30f; lrow[i] = 0.0f; }

  __shared__ __align__(16) _Float16 kts[64][64];           // 8 KB, [key][d]
  __shared__ __align__(16) _Float16 vts[64][64];           // 8 KB, [d][key]
  __shared__ __align__(16) _Float16 pls[4][16][64];        // 8 KB, wave-private P

  for (int t0 = 0; t0 < Seq; t0 += 64) {
    // ---- cooperative stage of K/V tiles into LDS (16B per thread-chunk) ----
    {
      const v8h* gk = (const v8h*)(Kb + (size_t)t0 * Dh);  // contiguous 8KB
      v8h* sk = (v8h*)&kts[0][0];
      #pragma unroll
      for (int j = 0; j < 4; ++j) sk[tid + 128 * j] = gk[tid + 128 * j];
      #pragma unroll
      for (int j = 0; j < 4; ++j) {
        const int c = tid + 128 * j;                       // 512 chunks of 16B
        const int r = c >> 3, cc = c & 7;
        ((v8h*)&vts[r][0])[cc] = ((const v8h*)(Vb + (size_t)r * Seq + t0))[cc];
      }
    }
    // prefetch next tile while this one is consumed
    {
      const int tn = (t0 + 64 < Seq) ? (t0 + 64) : 0;
      __builtin_prefetch(Kb + (size_t)tn * Dh + (size_t)tid * 64, 0, 1);
      __builtin_prefetch(Vb + (size_t)(tid >> 1) * Seq + tn + (tid & 1) * 32, 0, 1);
    }
    __syncthreads();

    // ---- S = Q K^T (base-2 logits; 1/sqrt(D)*log2e folded into Q) ----
    v8f s[4] = {v8f{}, v8f{}, v8f{}, v8f{}};
    #pragma unroll
    for (int n = 0; n < 4; ++n) {
      v16h bk0 = load_frag(&kts[n * 16][0],  64, lane);
      v16h bk1 = load_frag(&kts[n * 16][32], 64, lane);
      s[n] = wmma16(aq0, bk0, s[n]);
      s[n] = wmma16(aq1, bk1, s[n]);
    }

    // ---- online softmax (row M=i lives across lanes 0..15 of C VGPR i) ----
    float pr[4][8];
    #pragma unroll
    for (int i = 0; i < 8; ++i) {
      float mt = fmaxf(fmaxf(s[0][i], s[1][i]), fmaxf(s[2][i], s[3][i]));
      #pragma unroll
      for (int msk = 8; msk >= 1; msk >>= 1) mt = fmaxf(mt, __shfl_xor(mt, msk, 16));
      const float mnew = fmaxf(mrow[i], mt);
      const float corr = exp2f(mrow[i] - mnew);
      float rsum = 0.0f;
      #pragma unroll
      for (int n = 0; n < 4; ++n) {
        const float pv = exp2f(s[n][i] - mnew);
        pr[n][i] = pv;
        rsum += pv;
      }
      #pragma unroll
      for (int msk = 8; msk >= 1; msk >>= 1) rsum += __shfl_xor(rsum, msk, 16);
      lrow[i] = lrow[i] * corr + rsum;
      mrow[i] = mnew;
      #pragma unroll
      for (int n = 0; n < 4; ++n) acc[n][i] *= corr;
    }

    // ---- P (C layout) -> LDS -> A fragments ----
    #pragma unroll
    for (int n = 0; n < 4; ++n)
      #pragma unroll
      for (int i = 0; i < 8; ++i)
        pls[wave][hi * 8 + i][n * 16 + lo] = (_Float16)pr[n][i];
    __asm__ volatile("s_wait_dscnt 0x0" ::: "memory");     // wave-local RAW on LDS

    const v16h ap0 = load_frag(&pls[wave][0][0] +  0, 64, lane);
    const v16h ap1 = load_frag(&pls[wave][0][0] + 32, 64, lane);

    // ---- O += P V ----
    #pragma unroll
    for (int n = 0; n < 4; ++n) {
      v16h bv0 = load_frag(&vts[n * 16][0],  64, lane);
      v16h bv1 = load_frag(&vts[n * 16][32], 64, lane);
      acc[n] = wmma16(ap0, bv0, acc[n]);
      acc[n] = wmma16(ap1, bv1, acc[n]);
    }
    __syncthreads();      // all waves done with kts/vts before next overwrite
  }

  // ---- normalize, write ctx (heads concatenated along E) ----
  #pragma unroll
  for (int i = 0; i < 8; ++i) {
    const float inv = 1.0f / lrow[i];
    const size_t r = (size_t)b * Seq + q0 + hi * 8 + i;
    #pragma unroll
    for (int n = 0; n < 4; ++n)
      ctxh[r * Emb + h * Dh + n * 16 + lo] = (_Float16)(acc[n][i] * inv);
  }
}

// ---------------------------------------------------------------------------
// Output projection: ctx[Rows,Emb] x Wo + bo -> out (f32).
// grid (Rows/128, Emb/64), block 128; same 32x64-per-wave double-buffered loop.
// ---------------------------------------------------------------------------
__global__ void gemm_out_kernel(const _Float16* __restrict__ ctxh,
                                const _Float16* __restrict__ Wot,
                                const float* __restrict__ bo, float* __restrict__ out) {
  const int mt = blockIdx.x, nt = blockIdx.y;
  const int wave = threadIdx.x >> 5, lane = threadIdx.x & 31;
  const int lo = lane & 15, hi = lane >> 4;
  const int m0 = mt * 128 + wave * 32;
  const int n0 = nt * 64;

  const _Float16* A0 = ctxh + (size_t)m0 * Emb;
  const _Float16* A1 = A0 + (size_t)16 * Emb;
  const _Float16* Bm = Wot + (size_t)n0 * Emb;

  v16h a0 = load_frag(A0, Emb, lane);
  v16h a1 = load_frag(A1, Emb, lane);
  v16h b0 = load_frag(Bm + (size_t) 0 * Emb, Emb, lane);
  v16h b1 = load_frag(Bm + (size_t)16 * Emb, Emb, lane);
  v16h b2 = load_frag(Bm + (size_t)32 * Emb, Emb, lane);
  v16h b3 = load_frag(Bm + (size_t)48 * Emb, Emb, lane);

  v8f acc[2][4] = {};
  for (int k0 = 0; k0 < Emb; k0 += 32) {
    const int kn = (k0 + 32) & (Emb - 1);
    v16h a0n = load_frag(A0 + kn, Emb, lane);
    v16h a1n = load_frag(A1 + kn, Emb, lane);
    v16h bn0 = load_frag(Bm + (size_t) 0 * Emb + kn, Emb, lane);
    v16h bn1 = load_frag(Bm + (size_t)16 * Emb + kn, Emb, lane);
    v16h bn2 = load_frag(Bm + (size_t)32 * Emb + kn, Emb, lane);
    v16h bn3 = load_frag(Bm + (size_t)48 * Emb + kn, Emb, lane);

    acc[0][0] = wmma16(a0, b0, acc[0][0]);
    acc[1][0] = wmma16(a1, b0, acc[1][0]);
    acc[0][1] = wmma16(a0, b1, acc[0][1]);
    acc[1][1] = wmma16(a1, b1, acc[1][1]);
    acc[0][2] = wmma16(a0, b2, acc[0][2]);
    acc[1][2] = wmma16(a1, b2, acc[1][2]);
    acc[0][3] = wmma16(a0, b3, acc[0][3]);
    acc[1][3] = wmma16(a1, b3, acc[1][3]);

    a0 = a0n; a1 = a1n; b0 = bn0; b1 = bn1; b2 = bn2; b3 = bn3;
  }

  #pragma unroll
  for (int mi = 0; mi < 2; ++mi) {
    #pragma unroll
    for (int i = 0; i < 8; ++i) {
      const size_t row = (size_t)m0 + mi * 16 + hi * 8 + i;
      #pragma unroll
      for (int n = 0; n < 4; ++n) {
        const int col = n0 + n * 16 + lo;
        out[row * Emb + col] = acc[mi][n][i] + bo[col];
      }
    }
  }
}

// ---------------------------------------------------------------------------
extern "C" void kernel_launch(void* const* d_in, const int* in_sizes, int n_in,
                              void* d_out, int out_size, void* d_ws, size_t ws_size,
                              hipStream_t stream) {
  const float* x  = (const float*)d_in[0];
  const float* Wq = (const float*)d_in[1];
  const float* bq = (const float*)d_in[2];
  const float* Wk = (const float*)d_in[3];
  const float* bk = (const float*)d_in[4];
  const float* Wv = (const float*)d_in[5];
  const float* bv = (const float*)d_in[6];
  const float* Wo = (const float*)d_in[7];
  const float* bo = (const float*)d_in[8];
  float* out = (float*)d_out;

  char* ws = (char*)d_ws;
  _Float16* xh   = (_Float16*)(ws + (size_t) 0u);          // 16 MB  [Rows][Emb]
  _Float16* Wt   = (_Float16*)(ws + (size_t)(16u << 20));  //  6 MB  [3][H][D][E]
  _Float16* Wot  = (_Float16*)(ws + (size_t)(22u << 20));  //  2 MB  [E][E] transposed
  _Float16* Qf   = (_Float16*)(ws + (size_t)(24u << 20));  // 16 MB  [H][B][S][D]
  _Float16* Kf   = (_Float16*)(ws + (size_t)(40u << 20));  // 16 MB  [H][B][S][D]
  _Float16* Vt   = (_Float16*)(ws + (size_t)(56u << 20));  // 16 MB  [H][B][D][S]
  _Float16* ctxh = (_Float16*)(ws + (size_t)(72u << 20));  // 16 MB  [Rows][Emb]

  cvt_x_kernel   <<<(Rows * Emb) / 256, 256, 0, stream>>>(x, xh);
  cvt_wqkv_kernel<<<(3 * Hn * Dh * Emb) / 256, 256, 0, stream>>>(Wq, Wk, Wv, Wt);
  cvt_wo_kernel  <<<(Emb * Emb) / 256, 256, 0, stream>>>(Wo, Wot);

  gemm_qkv_kernel<<<dim3(Rows / 128, Hn, 3), 128, 0, stream>>>(xh, Wt, bq, bk, bv, Qf, Kf, Vt);
  attn_kernel    <<<dim3(Seq / 64, Hn, Bsz), 128, 0, stream>>>(Qf, Kf, Vt, ctxh);
  gemm_out_kernel<<<dim3(Rows / 128, Emb / 64), 128, 0, stream>>>(ctxh, Wot, bo, out);
}
